// ColorHistogramLoss_22187801051391
// MI455X (gfx1250) — compile-verified, matrix operand order checked by reading, not run on hardware
//
#include <hip/hip_runtime.h>
#include <math.h>

typedef __attribute__((ext_vector_type(16))) _Float16 v16h;
typedef __attribute__((ext_vector_type(2)))  __fp16   v2hf;   // cvt_pkrtz result type
typedef __attribute__((ext_vector_type(8)))  float    v8f;

#define NUM_BINS        64
#define BPP             64              // blocks per channel-plane
#define WAVES_PER_BLOCK 8
#define BLOCK_THREADS   (WAVES_PER_BLOCK * 32)
#define N_PLANES        12              // 2 tensors * (B=2 * C=3)

union B16 { v16h h; v2hf p[8]; };

// Wave broadcast: ds_bpermute wraps the index (only bits [6:2] used), so no
// explicit &31 mask is needed.
__device__ __forceinline__ float lane_bcast(float v, int srcLane) {
    return __int_as_float(
        __builtin_amdgcn_ds_bpermute(srcLane << 2, __float_as_int(v)));
}

// ---------------------------------------------------------------------------
__global__ __launch_bounds__(N_PLANES * NUM_BINS)
void zero_hist(float* __restrict__ hist) {
    hist[threadIdx.x] = 0.0f;
}

// ---------------------------------------------------------------------------
// Stage 1: soft histogram per channel-plane.
// Per 32-pixel tile: build four 32x16 f16 weight matrices (16 bins each) and
// reduce over pixels with v_wmma_f32_16x16x32_f16 against an all-ones A.
// With A == 1 the column sums are invariant to K-permutation, so lane l fills
// its 16 B-slots (column n = lane&15) with weights of pixels {l..l+15} and
// lane l^16 supplies the complementary 16 pixels (rotation broadcast).
// Weight math: w = exp2(K2*(u-c)^2) = exp2(K2*u^2 + m_g*u + b_g), evaluated
// with raw v_exp_f32 (f16 target flushes anything the raw exp would flush).
// ---------------------------------------------------------------------------
__global__ __launch_bounds__(BLOCK_THREADS)
void hist_kernel(const float* __restrict__ gen,
                 const float* __restrict__ tgt,
                 float* __restrict__ hist,
                 int P /* pixels per plane */) {
    const int plane = blockIdx.x / BPP;   // 0..11 : 0-5 generated, 6-11 target
    const int blk   = blockIdx.x % BPP;
    const float* base = (plane < 6) ? (gen + (size_t)plane * (size_t)P)
                                    : (tgt + (size_t)(plane - 6) * (size_t)P);

    const int lane = threadIdx.x & 31;
    const int wave = threadIdx.x >> 5;
    const int waveInPlane   = blk * WAVES_PER_BLOCK + wave;
    const int wavesPerPlane = BPP * WAVES_PER_BLOCK;

    // w = exp(-0.5*((u-c)/sigma)^2), sigma = 1.5/64, in exp2 space.
    const float sigma = 1.5f / 64.0f;
    const float K2    = -0.5f * 1.44269504088896341f / (sigma * sigma);

    // Per-lane quadratic coefficients for the 4 bin groups:
    // arg(u) = K2*u^2 + m_g*u + b_g
    float mg[4], bg[4];
#pragma unroll
    for (int g = 0; g < 4; ++g) {
        const float c = (float)(g * 16 + (lane & 15)) * (1.0f / 63.0f);
        mg[g] = -2.0f * K2 * c;
        bg[g] = K2 * c * c;
    }

    v16h ones;
#pragma unroll
    for (int j = 0; j < 16; ++j) ones[j] = (_Float16)1.0f;

    v8f acc[4];
#pragma unroll
    for (int g = 0; g < 4; ++g) acc[g] = (v8f){0, 0, 0, 0, 0, 0, 0, 0};

    const int groups = P >> 7;   // 128 pixels per wave iteration
    for (int grp = waveInPlane; grp < groups; grp += wavesPerPlane) {
        const float4 v4 = ((const float4*)base)[(size_t)grp * 32 + lane];
        float u[4];
        u[0] = fminf(fmaxf((v4.x + 1.0f) * 0.5f, 0.0f), 1.0f);
        u[1] = fminf(fmaxf((v4.y + 1.0f) * 0.5f, 0.0f), 1.0f);
        u[2] = fminf(fmaxf((v4.z + 1.0f) * 0.5f, 0.0f), 1.0f);
        u[3] = fminf(fmaxf((v4.w + 1.0f) * 0.5f, 0.0f), 1.0f);

#pragma unroll
        for (int t = 0; t < 4; ++t) {
            B16 bm[4];
#pragma unroll
            for (int j = 0; j < 8; ++j) {       // k pair = (2j, 2j+1)
                const float up0 = lane_bcast(u[t], lane + 2 * j);
                const float up1 = lane_bcast(u[t], lane + 2 * j + 1);
                const float q0  = K2 * up0 * up0;
                const float q1  = K2 * up1 * up1;
#pragma unroll
                for (int g = 0; g < 4; ++g) {
                    const float a0 = __builtin_fmaf(mg[g], up0, bg[g]) + q0;
                    const float a1 = __builtin_fmaf(mg[g], up1, bg[g]) + q1;
                    const float w0 = __builtin_amdgcn_exp2f(a0); // raw v_exp_f32
                    const float w1 = __builtin_amdgcn_exp2f(a1);
                    bm[g].p[j] = __builtin_amdgcn_cvt_pkrtz(w0, w1);
                }
            }
#pragma unroll
            for (int g = 0; g < 4; ++g)
                acc[g] = __builtin_amdgcn_wmma_f32_16x16x32_f16(
                    /*neg_a=*/false, ones, /*neg_b=*/false, bm[g].h,
                    /*c_mod=*/(short)0, acc[g],
                    /*reuse_a=*/false, /*reuse_b=*/false);
        }
    }

    // D row M=0 lives in VGPR0 of lanes 0..15 (N = lane): per-bin partial sums.
    __shared__ float sh[NUM_BINS];
    for (int i = threadIdx.x; i < NUM_BINS; i += BLOCK_THREADS) sh[i] = 0.0f;
    __syncthreads();
    if (lane < 16) {
#pragma unroll
        for (int g = 0; g < 4; ++g)
            atomicAdd(&sh[g * 16 + lane], acc[g][0]);   // ds_add_f32
    }
    __syncthreads();
    for (int i = threadIdx.x; i < NUM_BINS; i += BLOCK_THREADS)
        atomicAdd(&hist[plane * NUM_BINS + i], sh[i]);  // global fp32 atomic
}

// ---------------------------------------------------------------------------
// Stage 2: normalize each channel histogram and reduce L1 mean (384 elems).
// ---------------------------------------------------------------------------
__global__ __launch_bounds__(384)
void finalize_kernel(const float* __restrict__ hist, float* __restrict__ out) {
    __shared__ float sGen[6], sTgt[6], sLoss;
    const int i   = threadIdx.x;        // 0..383
    const int ch  = i >> 6;             // 0..5
    const int bin = i & 63;

    if (i < 6) { sGen[i] = 0.0f; sTgt[i] = 0.0f; }
    if (i == 0) sLoss = 0.0f;
    __syncthreads();

    const float g = hist[ch * NUM_BINS + bin];
    const float t = hist[(6 + ch) * NUM_BINS + bin];
    atomicAdd(&sGen[ch], g);
    atomicAdd(&sTgt[ch], t);
    __syncthreads();

    const float d = fabsf(g / (sGen[ch] + 1e-8f) - t / (sTgt[ch] + 1e-8f));
    atomicAdd(&sLoss, d);
    __syncthreads();

    if (i == 0) out[0] = sLoss * (1.0f / 384.0f);
}

// ---------------------------------------------------------------------------
extern "C" void kernel_launch(void* const* d_in, const int* in_sizes, int n_in,
                              void* d_out, int out_size, void* d_ws, size_t ws_size,
                              hipStream_t stream) {
    const float* gen = (const float*)d_in[0];
    const float* tgt = (const float*)d_in[1];
    float* out  = (float*)d_out;
    float* hist = (float*)d_ws;                 // 12*64 f32 = 3 KB scratch

    const int total = in_sizes[0];              // B*C*H*W = 2*3*512*512
    const int P     = total / 6;                // pixels per channel-plane

    zero_hist<<<1, N_PLANES * NUM_BINS, 0, stream>>>(hist);
    hist_kernel<<<N_PLANES * BPP, BLOCK_THREADS, 0, stream>>>(gen, tgt, hist, P);
    finalize_kernel<<<1, 384, 0, stream>>>(hist, out);
}
